// RealNVP_13821204759265
// MI455X (gfx1250) — compile-verified, matrix operand order checked by reading, not run on hardware
//
#include <hip/hip_runtime.h>
#include <math.h>

// ---------------------------------------------------------------------------
// RealNVP forward on MI455X (gfx1250), compute-bound (~412 GFLOP vs ~270 MB IO).
// f16 WMMA (v_wmma_f32_16x16x32_f16, f32 accum, inline-0 C). Weights repacked
// to B-fragment order in d_ws (L2-resident). Weight staging is a software
// pipeline over the 48-matrix schedule using the CDNA5 async-to-LDS path
// (ASYNCcnt): stage k+1's copy is issued before computing stage k, double-
// buffered in LDS, one barrier per stage. Biases are register-resident and
// fused into the epilogue (v_add + v_med3 relu). 8 waves / 128 rows per block.
// ---------------------------------------------------------------------------

typedef __attribute__((ext_vector_type(16))) _Float16 v16h;
typedef __attribute__((ext_vector_type(8)))  float    v8f;
typedef __attribute__((ext_vector_type(4)))  int      v4i;

#define BATCH 524288
#define ROWS_PER_BLOCK 128
#define THREADS 256
#define NWAVES 8
#define N_LAYERS 8

// packed weight sizes (halfs): W0 32x128, W1 128x128, W2 128x32
#define MLP_W_HALFS 24576                  // 4096 + 16384 + 4096
#define TOTAL_W_HALFS (16 * MLP_W_HALFS)   // 8 layers * 2 mlps
#define W_BYTES (TOTAL_W_HALFS * 2)        // 786432
#define MLP_B_FLOATS 288                   // 128 + 128 + 32

struct Ptrs96 { const float* p[96]; };

#if __has_builtin(__builtin_amdgcn_global_load_async_to_lds_b128)
#define HAS_ASYNC_LDS 1
#else
#define HAS_ASYNC_LDS 0
#endif

typedef __attribute__((address_space(1))) v4i  gv4i;
typedef __attribute__((address_space(3))) v4i  lv4i;

// ---- fragment helpers ------------------------------------------------------
// A-matrix 16-bit 16x32 layout (ISA 7.12.2): lane holds row M=lane%16;
// half h = 2*v + p maps to K = (v<4?0:16) + (lane<16?0:8) + (v&3)*2 + p.
// B fragments are stored pre-swizzled with the mirrored layout (lane = column).

__device__ __forceinline__ v16h load_frag(const _Float16* buf, int tile, int lane) {
    return *(const v16h*)(buf + tile * 512 + lane * 16);
}

// C = 0 -> backend uses the inline-0 SRC2 encoding (no splat movs).
__device__ __forceinline__ v8f wmma0(v16h a, v16h b) {
    v8f z = {};
    return __builtin_amdgcn_wmma_f32_16x16x32_f16(false, a, false, b, (short)0, z,
                                                  false, false);
}

__device__ __forceinline__ v8f wmma16(v16h a, v16h b, v8f c) {
    return __builtin_amdgcn_wmma_f32_16x16x32_f16(false, a, false, b, (short)0, c,
                                                  false, false);
}

// Store a 16x16 D-layout tile (vals[r] at row M=r+hi8, col ncol_base+lane16)
// into an A-fragment-ordered f16 LDS buffer (so next stage's A loads are
// contiguous v16h reads). ReLU via single v_med3_f32 (finite clamp keeps the
// med3 from being folded back into two maxes).
__device__ __forceinline__ void store_rows_as_afrag(_Float16* hw, const float* vals,
                                                    int ncol_base, bool relu, int lane) {
    const int lane16 = lane & 15;
    const int hi8    = (lane >> 4) << 3;
    const int K      = ncol_base + lane16;   // column index == next-stage K
    const int ktile  = K >> 5;
    const int kl     = K & 31;
    const int kk     = kl & 15;
    const int lane_hi = kk >> 3;
    const int v      = ((kl >> 4) << 2) | ((kk & 7) >> 1);
    const int p      = kk & 1;
    const int base   = ktile * 512 + lane_hi * 256 + (v << 1) + p;
#pragma unroll
    for (int r = 0; r < 8; ++r) {
        float x = vals[r];
        if (relu) x = __builtin_amdgcn_fmed3f(x, 0.0f, 3.0e38f);
        hw[base + (r + hi8) * 16] = (_Float16)x;
    }
}

// Issue-only async weight staging: global (L2-resident) -> LDS, no wait here.
// CDNA5 async direct-to-LDS path: no VGPR bounce, tracked by ASYNCcnt.
template <int NHALFS>
__device__ __forceinline__ void stage_issue(const _Float16* __restrict__ g,
                                            _Float16* s, int tid) {
    constexpr int n = NHALFS >> 3;             // 16-byte chunks
#if HAS_ASYNC_LDS
    gv4i* gp = (gv4i*)g;                       // AS1 int4* (builtin signature)
    lv4i* sp = (lv4i*)s;                       // AS3 int4*
#pragma unroll
    for (int i = 0; i < n / THREADS; ++i)
        __builtin_amdgcn_global_load_async_to_lds_b128(gp + tid + i * THREADS,
                                                       sp + tid + i * THREADS, 0, 0);
#else
    const uint4* gs = (const uint4*)g;
    uint4* ss = (uint4*)s;
#pragma unroll
    for (int i = 0; i < n / THREADS; ++i) ss[tid + i * THREADS] = gs[tid + i * THREADS];
#endif
}

// Publish the in-flight copy: wait for own async issues, then block barrier.
__device__ __forceinline__ void stage_publish() {
#if HAS_ASYNC_LDS
    asm volatile("s_wait_asynccnt 0" ::: "memory");
#endif
    __syncthreads();
}

// ---- prep: repack weights f32 -> f16 fragment order in d_ws ---------------
__global__ void prep_weights(Ptrs96 tab, _Float16* __restrict__ wsW) {
    const int mid = blockIdx.y;              // 0..47
    const int l = mid / 6, rem = mid % 6, m = rem / 3, j = rem % 3;
    const int Ks[3]   = {32, 128, 128};
    const int Ns[3]   = {128, 128, 32};
    const int joff[3] = {0, 4096, 20480};
    const int K = Ks[j], N = Ns[j];
    const int total = K * N;
    const int tid = blockIdx.x * 256 + threadIdx.x;
    if (tid >= total) return;
    const float* W = tab.p[(l * 2 + m) * 6 + j * 2];
    _Float16* dst = wsW + (l * 2 + m) * MLP_W_HALFS + joff[j];
    const int ktc    = K >> 5;
    const int tile   = tid >> 9;        // nt-major: tile = nt*ktc + kt
    const int within = tid & 511;
    const int lane   = within >> 4;
    const int h      = within & 15;
    const int v = h >> 1, p = h & 1;
    const int nt = tile / ktc, kt = tile % ktc;
    const int Klocal = ((v >= 4) ? 16 : 0) + ((lane >= 16) ? 8 : 0) + ((v & 3) << 1) + p;
    const int n = nt * 16 + (lane & 15);
    const int k = kt * 32 + Klocal;
    dst[tid] = (_Float16)W[k * N + n];
}

__global__ void prep_bias(Ptrs96 tab, float* __restrict__ wsB) {
    const int idx = blockIdx.x * 256 + threadIdx.x;
    if (idx >= 16 * MLP_B_FLOATS) return;
    const int lm = idx / MLP_B_FLOATS, r = idx % MLP_B_FLOATS;
    const int j = (r < 128) ? 0 : ((r < 256) ? 1 : 2);
    const int i = r - ((j == 0) ? 0 : ((j == 1) ? 128 : 256));
    wsB[idx] = tab.p[lm * 6 + j * 2 + 1][i];
}

// ---- one coupling layer ----------------------------------------------------
// Pipeline invariant at entry: this layer's s-MLP W0 is resident in wb[0] and
// a barrier has been passed since its copy completed. Each stage issues the
// next stage's copy before computing, so the async transfer overlaps WMMA.
// Buffer parity per stage s of mlp m is (3*m+s)&1 (6 stages/layer => repeats).
template <int XM0>
__device__ __forceinline__ void layer_step(int layer,
                                           float zt[4][8], float ld[8],
                                           _Float16* hw, _Float16 (*wb)[16384],
                                           const _Float16* __restrict__ wsW,
                                           const float* __restrict__ wsB,
                                           const _Float16* __restrict__ wnext,
                                           int tid, int lane, int lane16, int hi8) {
    constexpr int XU0 = 2 - XM0;

    // Build the xm A-fragment (16x32) once per layer; shared by s and t MLPs.
    store_rows_as_afrag(hw, zt[XM0 + 0], 0, false, lane);
    store_rows_as_afrag(hw, zt[XM0 + 1], 16, false, lane);
    asm volatile("s_wait_dscnt 0" ::: "memory");
    const v16h a_xm = load_frag(hw, 0, lane);

    v8f sacc[2], tacc[2];
    float b2r[2][2];                               // last-stage biases (s and t)
#pragma unroll
    for (int mlp = 0; mlp < 2; ++mlp) {
        const _Float16* wm = wsW + (layer * 2 + mlp) * MLP_W_HALFS;
        const float*    bm = wsB + (layer * 2 + mlp) * MLP_B_FLOATS;
        _Float16* const wb0 = wb[(3 * mlp + 0) & 1];   // this stage's resident buffer
        _Float16* const wb1 = wb[(3 * mlp + 1) & 1];
        _Float16* const wb2 = wb[(3 * mlp + 2) & 1];
        _Float16* const wb3 = wb[(3 * mlp + 3) & 1];   // next W0 target

        // Load this MLP's biases into registers in one clause; consumed only
        // in epilogues, so the L2 latency hides behind stage-1 compute.
        float b0r[8], b1r[8];
#pragma unroll
        for (int i = 0; i < 8; ++i) {
            b0r[i] = bm[i * 16 + lane16];
            b1r[i] = bm[128 + i * 16 + lane16];
        }
        b2r[mlp][0] = bm[256 + lane16];
        b2r[mlp][1] = bm[256 + 16 + lane16];

        // ---- stage 1: h1 = relu(xm @ W0 + b0)   (16x32)x(32x128); W0 in wb0
        stage_issue<16384>(wm + 4096, wb1, tid);       // prefetch W1
        v8f acc1[8];
#pragma unroll
        for (int nt = 0; nt < 8; ++nt) {
            const v16h b = load_frag(wb0, nt, lane);
            acc1[nt] = wmma0(a_xm, b);                 // inline-0 C
        }
#pragma unroll
        for (int nt = 0; nt < 8; ++nt) {
            float tmp[8];
#pragma unroll
            for (int r = 0; r < 8; ++r) tmp[r] = acc1[nt][r] + b0r[nt];
            store_rows_as_afrag(hw, tmp, nt * 16, true, lane);
        }
        asm volatile("s_wait_dscnt 0" ::: "memory");
        stage_publish();                               // W1 ready in wb1

        // ---- stage 2: h2 = relu(h1 @ W1 + b1)   (16x128)x(128x128)
        stage_issue<4096>(wm + 20480, wb2, tid);       // prefetch W2
        v8f acc2[8];
#pragma unroll
        for (int nt = 0; nt < 8; ++nt) {
#pragma unroll
            for (int kt = 0; kt < 4; ++kt) {
                const v16h a = load_frag(hw, kt, lane);
                const v16h b = load_frag(wb1, nt * 4 + kt, lane);
                acc2[nt] = (kt == 0) ? wmma0(a, b) : wmma16(a, b, acc2[nt]);
            }
        }
        // all reads of h1 done; overwrite hbuf with h2 (A-frag order)
#pragma unroll
        for (int nt = 0; nt < 8; ++nt) {
            float tmp[8];
#pragma unroll
            for (int r = 0; r < 8; ++r) tmp[r] = acc2[nt][r] + b1r[nt];
            store_rows_as_afrag(hw, tmp, nt * 16, true, lane);
        }
        asm volatile("s_wait_dscnt 0" ::: "memory");
        stage_publish();                               // W2 ready in wb2

        // ---- stage 3: o = h2 @ W2 + b2          (16x128)x(128x32)
        const _Float16* nxt = (mlp == 0) ? wm + MLP_W_HALFS : wnext;
        stage_issue<4096>(nxt, wb3, tid);              // prefetch next MLP/layer W0
        v8f* oacc = (mlp == 0) ? sacc : tacc;
#pragma unroll
        for (int nt = 0; nt < 2; ++nt) {
#pragma unroll
            for (int kt = 0; kt < 4; ++kt) {
                const v16h a = load_frag(hw, kt, lane);
                const v16h b = load_frag(wb2, nt * 4 + kt, lane);
                oacc[nt] = (kt == 0) ? wmma0(a, b) : wmma16(a, b, oacc[nt]);
            }
        }
        stage_publish();                               // next W0 ready
    } // mlp

    // coupling update: yu = xu * exp(s + b2s) + (t + b2t) ; log_det += sum(s)
#pragma unroll
    for (int ct = 0; ct < 2; ++ct)
#pragma unroll
        for (int r = 0; r < 8; ++r) {
            const float sv = sacc[ct][r] + b2r[0][ct];
            const float tv = tacc[ct][r] + b2r[1][ct];
            zt[XU0 + ct][r] = zt[XU0 + ct][r] * __expf(sv) + tv;
            ld[r] += sv;
        }
}

// ---- main fused flow kernel ------------------------------------------------
__global__ __launch_bounds__(THREADS)
void realnvp_main(const float* __restrict__ x,
                  const _Float16* __restrict__ wsW,
                  const float* __restrict__ wsB,
                  float* __restrict__ out) {
    __shared__ __align__(32) _Float16 wb[2][16384];         // 2x32KB: weight ping-pong
    __shared__ __align__(32) _Float16 hbuf[NWAVES][2048];   // 4KB per wave: activations

    const int tid    = threadIdx.x;
    const int lane   = tid & 31;
    const int wid    = tid >> 5;
    const int lane16 = lane & 15;
    const int hi8    = (lane >> 4) << 3;
    const long rowbase = (long)blockIdx.x * ROWS_PER_BLOCK + wid * 16;

    // Pipeline prologue: bring layer0 s-MLP W0 into wb[0].
    stage_issue<4096>(wsW, wb[0], tid);

    // z in registers, D layout: zt[ct][r] = z[rowbase+r+hi8][ct*16+lane16]
    float zt[4][8];
#pragma unroll
    for (int ct = 0; ct < 4; ++ct)
#pragma unroll
        for (int r = 0; r < 8; ++r)
            zt[ct][r] = x[(rowbase + r + hi8) * 64 + ct * 16 + lane16];

    float ld[8];
#pragma unroll
    for (int r = 0; r < 8; ++r) ld[r] = 0.0f;

    _Float16* hw = hbuf[wid];
    stage_publish();                                   // W0(layer0,s) resident

#pragma unroll 1
    for (int pair = 0; pair < N_LAYERS / 2; ++pair) {
        const _Float16* wn0 = wsW + (2 * pair + 1) * 2 * MLP_W_HALFS;
        const _Float16* wn1 = (pair == 3) ? wsW : wsW + (2 * pair + 2) * 2 * MLP_W_HALFS;
        layer_step<0>(2 * pair + 0, zt, ld, hw, wb, wsW, wsB, wn0, tid, lane, lane16, hi8);
        layer_step<2>(2 * pair + 1, zt, ld, hw, wb, wsW, wsB, wn1, tid, lane, lane16, hi8);
    }

    // ---- outputs: z (row-major), then log_det ------------------------------
#pragma unroll
    for (int ct = 0; ct < 4; ++ct)
#pragma unroll
        for (int r = 0; r < 8; ++r)
            out[(rowbase + r + hi8) * 64 + ct * 16 + lane16] = zt[ct][r];

    float* ldout = out + (long)BATCH * 64;
#pragma unroll
    for (int r = 0; r < 8; ++r) {
        float v = ld[r];
        v += __shfl_xor(v, 1, 32);
        v += __shfl_xor(v, 2, 32);
        v += __shfl_xor(v, 4, 32);
        v += __shfl_xor(v, 8, 32);     // reduce over the 16-lane column group
        if (lane16 == 0) ldout[rowbase + r + hi8] = v;
    }
}

// ---------------------------------------------------------------------------
extern "C" void kernel_launch(void* const* d_in, const int* in_sizes, int n_in,
                              void* d_out, int out_size, void* d_ws, size_t ws_size,
                              hipStream_t stream) {
    Ptrs96 tab;
    for (int i = 0; i < 96; ++i) tab.p[i] = (const float*)d_in[1 + i];

    _Float16* wsW = (_Float16*)d_ws;
    float*    wsB = (float*)((char*)d_ws + W_BYTES);

    prep_weights<<<dim3(64, 48), 256, 0, stream>>>(tab, wsW);
    prep_bias<<<dim3(18, 1), 256, 0, stream>>>(tab, wsB);

    realnvp_main<<<BATCH / ROWS_PER_BLOCK, THREADS, 0, stream>>>(
        (const float*)d_in[0], wsW, wsB, (float*)d_out);
}
